// QLSTMGen195_65481071404263
// MI455X (gfx1250) — compile-verified, matrix operand order checked by reading, not run on hardware
//
#include <hip/hip_runtime.h>
#include <hip/hip_bf16.h>

#define T_STEPS 512
#define BATCH   256
#define DIN     256
#define HID     64
#define N4H     256   // 4*HID
#define DH      320   // DIN + HID

typedef __attribute__((ext_vector_type(16))) __bf16 v16bf;
typedef __attribute__((ext_vector_type(8)))  float  v8f;

union FragAB { v16bf v; unsigned u[8]; };

__device__ __forceinline__ unsigned short f2bf(float f) {
  union { float f; unsigned u; } x; x.f = f;
  unsigned r = x.u + 0x7FFFu + ((x.u >> 16) & 1u);   // round-to-nearest-even
  return (unsigned short)(r >> 16);
}

// Even K-base for (vgpr v, lane) in the 16-bit 16x32 A/B fragment layout:
// lanes 0-15: v0..3 -> K=0..7, v4..7 -> K=16..23 ; lanes 16-31: +8.
__device__ __forceinline__ int frag_k0(int v, int lane) {
  return ((v >= 4) ? 16 : 0) + ((v & 3) * 2) + ((lane >= 16) ? 8 : 0);
}

__device__ __forceinline__ v8f wmma_bf16(v16bf a, v16bf b, v8f c) {
  return __builtin_amdgcn_wmma_f32_16x16x32_bf16(false, a, false, b, (short)0, c,
                                                 false, false);
}

// ---------------- Weight packing: fragment-native [frag][lane][16 bf16] ------

// Wx: rows k in [0,256) of W_gates, columns n = g*64+h in [0,256). 128 frags.
__global__ void __launch_bounds__(256) pack_wx(const float* __restrict__ W,
                                               unsigned short* __restrict__ dst) {
  int t = blockIdx.x * 256 + threadIdx.x;   // frag*32 + lane, 4096 threads
  int lane = t & 31, frag = t >> 5;         // frag = kb*16 + nt
  int kb = frag >> 4, nt = frag & 15;
  unsigned short* out = dst + ((size_t)frag * 32 + lane) * 16;
  int n = nt * 16 + (lane & 15);
  int g = n >> 6, hcol = n & 63;
  for (int s = 0; s < 16; ++s) {
    int k = kb * 32 + frag_k0(s >> 1, lane) + (s & 1);
    out[s] = f2bf(W[(size_t)g * DH * HID + (size_t)k * HID + hcol]);
  }
}

// Wh: rows k in [0,64) mapped to W_gates rows 256+k. 32 frags (kb*16+nt).
__global__ void __launch_bounds__(256) pack_wh(const float* __restrict__ W,
                                               unsigned short* __restrict__ dst) {
  int t = blockIdx.x * 256 + threadIdx.x;   // 1024 threads
  int lane = t & 31, frag = t >> 5;
  int kb = frag >> 4, nt = frag & 15;
  unsigned short* out = dst + ((size_t)frag * 32 + lane) * 16;
  int n = nt * 16 + (lane & 15);
  int g = n >> 6, hcol = n & 63;
  for (int s = 0; s < 16; ++s) {
    int k = kb * 32 + frag_k0(s >> 1, lane) + (s & 1);
    out[s] = f2bf(W[(size_t)g * DH * HID + (size_t)(DIN + k) * HID + hcol]);
  }
}

// Wq: per-gate [64,64]; frag id = ((g*4 + j)*2 + kb). 32 frags.
__global__ void __launch_bounds__(256) pack_wq(const float* __restrict__ W,
                                               unsigned short* __restrict__ dst) {
  int t = blockIdx.x * 256 + threadIdx.x;   // 1024 threads
  int lane = t & 31, frag = t >> 5;
  int kb = frag & 1, gj = frag >> 1;
  int g = gj >> 2, j = gj & 3;
  unsigned short* out = dst + ((size_t)frag * 32 + lane) * 16;
  int ncol = j * 16 + (lane & 15);
  for (int s = 0; s < 16; ++s) {
    int k = kb * 32 + frag_k0(s >> 1, lane) + (s & 1);
    out[s] = f2bf(W[(size_t)g * HID * HID + (size_t)k * HID + ncol]);
  }
}

// ---------------- Phase 1: Zx = x @ Wx + b_gates  (fragment-packed output) ---

__global__ void __launch_bounds__(256) xproj_kernel(
    const float* __restrict__ x, const float* __restrict__ b_gates,
    const unsigned short* __restrict__ WxP, float* __restrict__ Zx) {
  __shared__ __align__(16) unsigned short xbf[16 * DIN];   // 8 KB bf16 tile
  int tid = threadIdx.x;
  int mt = blockIdx.x;                       // 0..8191  (= t*16 + btile)
  size_t r0 = (size_t)mt * 16;
  for (int i = tid; i < 16 * DIN; i += 256) {
    int row = i >> 8, d = i & 255;
    xbf[i] = f2bf(x[(r0 + row) * DIN + d]);
  }
  __syncthreads();

  int wave = tid >> 5, lane = tid & 31;
  int m = lane & 15;
  FragAB a[8];                               // K = 256 -> 8 A fragments
  for (int kb = 0; kb < 8; ++kb)
    for (int v = 0; v < 8; ++v)
      a[kb].u[v] = *(const unsigned*)(const void*)
          &xbf[m * DIN + kb * 32 + frag_k0(v, lane)];

  for (int nn = 0; nn < 2; ++nn) {
    int nt = wave * 2 + nn;                  // each wave: 2 of 16 N-tiles
    float bv = b_gates[nt * 16 + (lane & 15)];  // col = g*64+h = flat index
    v8f acc = {bv, bv, bv, bv, bv, bv, bv, bv};
    for (int kb = 0; kb < 8; ++kb) {
      v16bf bfrag = *(const v16bf*)(const void*)
          (WxP + ((size_t)(kb * 16 + nt) * 32 + lane) * 16);
      acc = wmma_bf16(a[kb].v, bfrag, acc);
    }
    *(v8f*)(void*)(Zx + (((size_t)mt * 16 + nt) * 32 + lane) * 8) = acc;
  }
}

// ---------------- Phase 2: recurrence (16 independent batch tiles) ----------

__global__ void __launch_bounds__(256) recurrent_kernel(
    const float* __restrict__ bq, const unsigned short* __restrict__ WhP,
    const unsigned short* __restrict__ WqP, const float* __restrict__ Zx,
    float* __restrict__ out) {
  __shared__ __align__(16) unsigned short h_lds[16 * HID];  // bf16 h state
  __shared__ __align__(16) unsigned short z_lds[16 * N4H];  // bf16 gate preact
  __shared__ __align__(16) float q_lds[4][16 * HID];        // f32 q per gate
  __shared__ __align__(16) float c_lds[16 * HID];
  __shared__ __align__(16) float hf_lds[16 * HID];

  int tid = threadIdx.x, wave = tid >> 5, lane = tid & 31;
  int blk = blockIdx.x;                      // batch tile 0..15
  for (int e = tid; e < 16 * HID; e += 256) { h_lds[e] = 0; c_lds[e] = 0; }

  // Preload constant B fragments (held in VGPRs across all 512 steps).
  FragAB wh[2][2];                           // [nn][kb]; nt = wave + nn*8
  for (int nn = 0; nn < 2; ++nn) {
    int nt = wave + nn * 8;
    for (int kb = 0; kb < 2; ++kb)
      wh[nn][kb].v = *(const v16bf*)(const void*)
          (WhP + ((size_t)(kb * 16 + nt) * 32 + lane) * 16);
  }
  FragAB wq[2][2]; float bqv[2];
  int tiles[2] = {wave, wave + 8};           // q-tile t2: g = t2>>2, j = t2&3
  for (int tt = 0; tt < 2; ++tt) {
    int t2 = tiles[tt], g = t2 >> 2, j = t2 & 3;
    for (int kb = 0; kb < 2; ++kb)
      wq[tt][kb].v = *(const v16bf*)(const void*)
          (WqP + ((size_t)((g * 4 + j) * 2 + kb) * 32 + lane) * 16);
    bqv[tt] = bq[g * 64 + j * 16 + (lane & 15)];
  }

  int m = lane & 15;
  int mhi = (lane >= 16) ? 8 : 0;

  for (int t = 0; t < T_STEPS; ++t) {
    __syncthreads();                         // h_lds / z_lds / q_lds settled

    // Build A (h) fragments: 16x64 bf16 from LDS, packed-pair b32 reads.
    FragAB ah[2];
    for (int kb = 0; kb < 2; ++kb)
      for (int v = 0; v < 8; ++v)
        ah[kb].u[v] = *(const unsigned*)(const void*)
            &h_lds[m * HID + kb * 32 + frag_k0(v, lane)];

    const float* zbase = Zx + ((size_t)(t * 16 + blk) * 16) * 32 * 8;
    if (t + 1 < T_STEPS)                     // prefetch part of Zx slab of Mtile t*16+blk+1
      __builtin_prefetch(zbase + (size_t)16 * 32 * 8 + (size_t)lane * 64, 0, 0);

    // z = Zx + h @ Wh   (each wave: 2 of 16 N-tiles)
    for (int nn = 0; nn < 2; ++nn) {
      int nt = wave + nn * 8;
      v8f acc = *(const v8f*)(const void*)(zbase + ((size_t)nt * 32 + lane) * 8);
      acc = wmma_bf16(ah[0].v, wh[nn][0].v, acc);
      acc = wmma_bf16(ah[1].v, wh[nn][1].v, acc);
      int ncol = nt * 16 + (lane & 15);
      for (int v = 0; v < 8; ++v)
        z_lds[(v + mhi) * N4H + ncol] = f2bf(acc[v]);
    }
    __syncthreads();

    // q[g] = z[g] @ Wq[g] + bq[g]   (each wave: 2 of 16 q-tiles)
    for (int tt = 0; tt < 2; ++tt) {
      int t2 = tiles[tt], g = t2 >> 2, j = t2 & 3;
      float bv = bqv[tt];
      v8f qa = {bv, bv, bv, bv, bv, bv, bv, bv};
      for (int kb = 0; kb < 2; ++kb) {
        FragAB az;
        for (int v = 0; v < 8; ++v)
          az.u[v] = *(const unsigned*)(const void*)
              &z_lds[m * N4H + g * 64 + kb * 32 + frag_k0(v, lane)];
        qa = wmma_bf16(az.v, wq[tt][kb].v, qa);
      }
      int ncol = j * 16 + (lane & 15);
      for (int v = 0; v < 8; ++v)
        q_lds[g][(v + mhi) * HID + ncol] = qa[v];
    }
    __syncthreads();

    // LSTM elementwise on the 16x64 tile; write h to output (coalesced).
    for (int e = tid; e < 16 * HID; e += 256) {
      float xf = q_lds[0][e], xi = q_lds[1][e];
      float xg = q_lds[2][e], xo = q_lds[3][e];
      float fg = 1.f / (1.f + __expf(-xf));
      float ig = 1.f / (1.f + __expf(-xi));
      float gg = tanhf(xg);
      float og = 1.f / (1.f + __expf(-xo));
      float c  = fg * c_lds[e] + ig * gg;
      c_lds[e] = c;
      float h  = og * tanhf(c);
      hf_lds[e] = h;
      h_lds[e]  = f2bf(h);
      out[(size_t)t * BATCH * HID + (size_t)(blk * 16 + (e >> 6)) * HID + (e & 63)] = h;
    }
  }
  __syncthreads();
  // Final hx, cx
  size_t base = (size_t)T_STEPS * BATCH * HID;
  for (int e = tid; e < 16 * HID; e += 256) {
    size_t off = (size_t)(blk * 16 + (e >> 6)) * HID + (e & 63);
    out[base + off] = hf_lds[e];
    out[base + (size_t)BATCH * HID + off] = c_lds[e];
  }
}

// ---------------- Host-side launch ------------------------------------------

extern "C" void kernel_launch(void* const* d_in, const int* in_sizes, int n_in,
                              void* d_out, int out_size, void* d_ws, size_t ws_size,
                              hipStream_t stream) {
  const float* x       = (const float*)d_in[0];   // [512,256,256]
  const float* W_gates = (const float*)d_in[1];   // [4,320,64]
  const float* b_gates = (const float*)d_in[2];   // [4,64]
  const float* Wq      = (const float*)d_in[3];   // [4,64,64]
  const float* bq      = (const float*)d_in[4];   // [4,64]
  float* out = (float*)d_out;                     // [512*256*64 + 2*256*64]

  // Workspace layout (bytes):
  //   [0, 131072)          Wx packed bf16 (128 frags * 1024B)
  //   [131072, 163840)     Wh packed bf16 (32 frags)
  //   [163840, 196608)     Wq packed bf16 (32 frags)
  //   [196608, +134217728) Zx f32 fragment-packed [8192 Mtiles][16 Nt][32][8]
  char* ws = (char*)d_ws;
  unsigned short* WxP = (unsigned short*)(ws);
  unsigned short* WhP = (unsigned short*)(ws + 131072);
  unsigned short* WqP = (unsigned short*)(ws + 163840);
  float*          Zx  = (float*)(ws + 196608);

  pack_wx<<<16, 256, 0, stream>>>(W_gates, WxP);
  pack_wh<<<4, 256, 0, stream>>>(W_gates, WhP);
  pack_wq<<<4, 256, 0, stream>>>(Wq, WqP);
  xproj_kernel<<<(T_STEPS * BATCH) / 16, 256, 0, stream>>>(x, b_gates, WxP, Zx);
  recurrent_kernel<<<BATCH / 16, 256, 0, stream>>>(bq, WhP, WqP, Zx, out);
}
